// VQEMAQuantizer_81535659147901
// MI455X (gfx1250) — compile-verified, matrix operand order checked by reading, not run on hardware
//
#include <hip/hip_runtime.h>
#include <hip/hip_bf16.h>
#include <math.h>

typedef __attribute__((ext_vector_type(16))) _Float16 v16h;
typedef __attribute__((ext_vector_type(8)))  float    v8f;

#define VQ_K   1024
#define VQ_D   64
#define VQ_N   131072      // 32*64*64
#define VQ_HW  4096        // 64*64
#define VQ_DECAY 0.99f
#define VQ_OMD   0.01f
#define VQ_EPS   1e-5f
#define VQ_CC    0.25f

// ---------------------------------------------------------------- zero scratch
__global__ __launch_bounds__(256) void vq_zero_kernel(float* __restrict__ counts,
                                                      float* __restrict__ dw,
                                                      float* __restrict__ scal) {
  int i = blockIdx.x * 256 + threadIdx.x;            // 65536 threads
  if (i < VQ_K) counts[i] = 0.0f;
  if (i < VQ_K * VQ_D) dw[i] = 0.0f;
  if (i < 16) scal[i] = 0.0f;
}

// ------------------------------------------------ pack codebook into B layout
// bpack record (chunk c, half h, lane l) = 16 f16 (32B) = exactly the v16h this
// lane feeds v_wmma_f32_16x16x32_f16 as B:
//   column N = c*16 + (l&15); K = h*32 + ((l>=16)?8:0) + (t<8 ? t : t+8)
// cnp[k] = 1 + ||e_k||^2  (bias makes all distances positive for u32-key argmin)
__global__ __launch_bounds__(256) void vq_prep_kernel(const float* __restrict__ emb,
                                                      _Float16* __restrict__ bpack,
                                                      float* __restrict__ cnp) {
  int i = blockIdx.x * 256 + threadIdx.x;            // 65536 threads
  {
    int t = i & 15;
    int l = (i >> 4) & 31;
    int h = (i >> 9) & 1;
    int c = i >> 10;
    int j = c * 16 + (l & 15);
    int k = h * 32 + ((l >> 4) << 3) + ((t < 8) ? t : t + 8);
    bpack[i] = (_Float16)emb[j * VQ_D + k];
  }
  if (i < VQ_K) {
    float s = 0.0f;
    #pragma unroll 8
    for (int d = 0; d < VQ_D; ++d) { float e = emb[i * VQ_D + d]; s += e * e; }
    cnp[i] = 1.0f + s;
  }
}

// ------------------------------------------------ distance argmin (WMMA core)
// 512 blocks x 256 thr (8 waves). Each wave owns 32 rows (two 16-row M tiles,
// A held in registers across the whole code stream). B tiles are double-
// buffered in LDS via async global->LDS DMA (in-order completion lets
// s_wait_asynccnt 4 retire only the older batch while the next is in flight).
// The dead A-staging buffer is overlaid by the B double buffer.
__global__ __launch_bounds__(256) void vq_argmin_kernel(const float* __restrict__ z,
                                                        const _Float16* __restrict__ bpack,
                                                        const float* __restrict__ cnp,
                                                        int* __restrict__ idx_out,
                                                        float* __restrict__ idx_out_f) {
  // [0,36864): A staging (256 rows x 72 f16, stride 144B = 36dw -> the 16-lane
  //            b128 reads are bank-conflict-free), later reused as 2x16KB B buf.
  // [36864,40960): 1 + ||e||^2 table.
  __shared__ __align__(16) char smem[40960];
  _Float16* aL  = (_Float16*)smem;
  _Float16* bL  = (_Float16*)smem;          // overlays aL after A is in regs
  float*    cnL = (float*)(smem + 36864);

  const int tid = threadIdx.x;
  const int n0  = blockIdx.x * 256;
  const int b   = n0 >> 12;            // tile never crosses b (256 | 4096)
  const int rem = n0 & 4095;

  for (int i = tid; i < VQ_K; i += 256) cnL[i] = cnp[i];

  // Stage z tile: coalesced global reads, permuted f16 stores into A layout.
  const float* zb = z + (size_t)b * VQ_D * VQ_HW;
  #pragma unroll 4
  for (int it = 0; it < 64; ++it) {
    int i  = it * 256 + tid;
    int d  = i >> 8;                   // 0..63
    int nl = i & 255;                  // local row
    float v = zb[(size_t)d * VQ_HW + rem + nl];
    int kp = d & 31, h = d >> 5;
    int q  = (kp >> 3) & 1;
    int t  = (kp & 7) + ((kp >> 4) << 3);
    aL[nl * 72 + (h * 2 + q) * 16 + t] = (_Float16)v;
  }
  __syncthreads();

  const int lane = tid & 31;
  const int w    = tid >> 5;           // wave id 0..7
  const int lrow = lane & 15;
  const int half = lane >> 4;          // 0: M rows 0..7 | 1: rows 8..15
  const int row0 = w * 32 + lrow;      // tile 0 A row
  const int row1 = row0 + 16;          // tile 1 A row

  // Hoist A into registers: invariant across all 64 code chunks.
  const v16h a00 = *(const v16h*)(aL + row0 * 72 + (0 * 2 + half) * 16);
  const v16h a01 = *(const v16h*)(aL + row0 * 72 + (1 * 2 + half) * 16);
  const v16h a10 = *(const v16h*)(aL + row1 * 72 + (0 * 2 + half) * 16);
  const v16h a11 = *(const v16h*)(aL + row1 * 72 + (1 * 2 + half) * 16);
  __syncthreads();                     // aL dead; B buffers may overwrite it

  unsigned int bk0[8], bk1[8];
  #pragma unroll
  for (int r = 0; r < 8; ++r) { bk0[r] = 0xFFFFFFFFu; bk1[r] = 0xFFFFFFFFu; }

  const unsigned int ldsb = (unsigned int)(uintptr_t)(&bL[0]);
  const char* bsrc = (const char*)bpack;

  // 4 async b128 loads per thread = one 16KB chunk-group of packed B.
  auto issue_cg = [&](int cg, int p) {
    #pragma unroll
    for (int jj = 0; jj < 4; ++jj) {
      int i = jj * 256 + tid;                          // 16B record index
      unsigned int voff = (unsigned int)(cg * 16384 + i * 16);
      unsigned int dst  = ldsb + (unsigned int)(p * 16384 + i * 16);
      asm volatile("global_load_async_to_lds_b128 %0, %1, %2 offset:0"
                   :: "v"(dst), "v"(voff), "s"(bsrc) : "memory");
    }
  };

  issue_cg(0, 0);
  issue_cg(1, 1);
  asm volatile("s_wait_asynccnt 0x4" ::: "memory");   // buf0 landed; buf1 in flight
  __syncthreads();

  #pragma unroll 1
  for (int cg = 0; cg < 8; ++cg) {     // 8 chunk-groups of 128 codes
    const int p = cg & 1;
    const _Float16* bbuf = bL + p * 8192;

    #pragma unroll 2
    for (int cl = 0; cl < 8; ++cl) {   // 16 codes per chunk
      const v16h b0 = *(const v16h*)(bbuf + (cl * 2 + 0) * 512 + lane * 16);
      const v16h b1 = *(const v16h*)(bbuf + (cl * 2 + 1) * 512 + lane * 16);
      v8f acc0 = {}, acc1 = {};
      acc0 = __builtin_amdgcn_wmma_f32_16x16x32_f16(false, a00, false, b0, (short)0, acc0, false, false);
      acc1 = __builtin_amdgcn_wmma_f32_16x16x32_f16(false, a10, false, b0, (short)0, acc1, false, false);
      acc0 = __builtin_amdgcn_wmma_f32_16x16x32_f16(false, a01, false, b1, (short)0, acc0, false, false);
      acc1 = __builtin_amdgcn_wmma_f32_16x16x32_f16(false, a11, false, b1, (short)0, acc1, false, false);

      const int j = (cg * 8 + cl) * 16 + lrow;
      const float cn = cnL[j];
      const unsigned int jv = (unsigned int)j;
      // key = (float_bits(1 + ||e||^2 - 2 z.e) & ~0x3FF) | idx : positive floats
      // order like u32, so v_min_u32 does (dist, idx) lexicographic argmin.
      #pragma unroll
      for (int r = 0; r < 8; ++r) {
        float d0 = fmaf(-2.0f, acc0[r], cn);
        float d1 = fmaf(-2.0f, acc1[r], cn);
        unsigned int k0 = (__float_as_uint(d0) & 0xFFFFFC00u) | jv;
        unsigned int k1 = (__float_as_uint(d1) & 0xFFFFFC00u) | jv;
        bk0[r] = min(bk0[r], k0);
        bk1[r] = min(bk1[r], k1);
      }
    }

    if (cg == 7) break;
    __syncthreads();                   // all waves done reading bbuf
    if (cg < 6) {
      issue_cg(cg + 2, p);             // refill the buffer we just drained
      asm volatile("s_wait_asynccnt 0x4" ::: "memory");  // cg+1 done, cg+2 in flight
    } else {
      asm volatile("s_wait_asynccnt 0x0" ::: "memory");  // final chunk-group done
    }
    __syncthreads();
  }

  // Per-row argmin across the 16 lanes of each half; u32-min keeps the
  // smallest index on distance ties (matches jnp.argmin first-occurrence).
  #pragma unroll
  for (int r = 0; r < 8; ++r) {
    unsigned int k0 = bk0[r], k1 = bk1[r];
    #pragma unroll
    for (int m = 8; m >= 1; m >>= 1) {
      k0 = min(k0, (unsigned int)__shfl_xor((int)k0, m, 16));
      k1 = min(k1, (unsigned int)__shfl_xor((int)k1, m, 16));
    }
    if (lrow == r) {
      int g0 = n0 + w * 32 + half * 8 + r;
      int g1 = g0 + 16;
      int i0 = (int)(k0 & 1023u);
      int i1 = (int)(k1 & 1023u);
      idx_out[g0] = i0;  idx_out_f[g0] = (float)i0;
      idx_out[g1] = i1;  idx_out_f[g1] = (float)i1;
    }
  }
}

// ------------------------------------------------------- scatter counts / dw
__global__ __launch_bounds__(256) void vq_scatter_kernel(const float* __restrict__ z,
                                                         const int* __restrict__ idx,
                                                         float* __restrict__ counts,
                                                         float* __restrict__ dw) {
  int n = blockIdx.x * 256 + threadIdx.x;
  int b = n >> 12, rem = n & 4095;
  int code = idx[n];
  atomicAdd(&counts[code], 1.0f);
  const float* zb = z + (size_t)b * VQ_D * VQ_HW + rem;
  float* dwr = dw + code * VQ_D;
  #pragma unroll 8
  for (int d = 0; d < VQ_D; ++d) atomicAdd(&dwr[d], zb[(size_t)d * VQ_HW]);
}

// ----------------------------------------- z_q gather + commitment loss sum
__global__ __launch_bounds__(256) void vq_zq_loss_kernel(const float* __restrict__ z,
                                                         const float* __restrict__ emb,
                                                         const int* __restrict__ idx,
                                                         float* __restrict__ zq,
                                                         float* __restrict__ loss_acc) {
  __shared__ float red[256];
  int tid = threadIdx.x;
  int n = blockIdx.x * 256 + tid;
  int b = n >> 12, rem = n & 4095;
  int code = idx[n];
  const float* zb = z  + (size_t)b * VQ_D * VQ_HW + rem;
  float*       ob = zq + (size_t)b * VQ_D * VQ_HW + rem;
  const float* e  = emb + code * VQ_D;
  float s = 0.0f;
  #pragma unroll 8
  for (int d = 0; d < VQ_D; ++d) {
    float zv = zb[(size_t)d * VQ_HW];
    float q  = e[d];
    float df = zv - q;
    s += df * df;
    ob[(size_t)d * VQ_HW] = q;   // straight-through output == quantized value
  }
  red[tid] = s;
  __syncthreads();
  #pragma unroll
  for (int off = 128; off > 0; off >>= 1) {
    if (tid < off) red[tid] += red[tid + off];
    __syncthreads();
  }
  if (tid == 0) atomicAdd(loss_acc, red[0]);
}

// -------------------------------------------------- n / entropy / usage stats
__global__ __launch_bounds__(1024) void vq_stats_kernel(const float* __restrict__ ema_cs,
                                                        const float* __restrict__ counts,
                                                        float* __restrict__ scal) {
  __shared__ float s_n[1024], s_e[1024], s_u[1024];
  int k = threadIdx.x;
  float cnt = counts[k];
  float raw = ema_cs[k] * VQ_DECAY + VQ_OMD * cnt;
  float p   = cnt * (1.0f / (float)VQ_N);   // sum(counts) == N exactly
  s_n[k] = raw;
  s_e[k] = (cnt > 0.0f) ? (-p * logf(p)) : 0.0f;
  s_u[k] = (cnt > 0.0f) ? 1.0f : 0.0f;
  __syncthreads();
  for (int off = 512; off > 0; off >>= 1) {
    if (k < off) { s_n[k] += s_n[k+off]; s_e[k] += s_e[k+off]; s_u[k] += s_u[k+off]; }
    __syncthreads();
  }
  if (k == 0) { scal[0] = s_n[0]; scal[1] = s_e[0]; scal[2] = s_u[0]; }
}

// --------------------------------------------------------- EMA finalize
__global__ __launch_bounds__(256) void vq_final_kernel(const float* __restrict__ ema_cs,
                                                       const float* __restrict__ ema_w,
                                                       const float* __restrict__ counts,
                                                       const float* __restrict__ dw,
                                                       const float* __restrict__ scal,
                                                       float* __restrict__ out_emb,
                                                       float* __restrict__ out_cs,
                                                       float* __restrict__ out_emaw,
                                                       float* __restrict__ out_scalars) {
  int i = blockIdx.x * 256 + threadIdx.x;   // 65536 = K*D
  int k = i >> 6, d = i & 63;
  float nval = scal[0];
  float raw  = ema_cs[k] * VQ_DECAY + VQ_OMD * counts[k];
  float ncs  = (raw + VQ_EPS) / (nval + (float)VQ_K * VQ_EPS) * nval;
  float nw   = ema_w[i] * VQ_DECAY + VQ_OMD * dw[i];
  out_emaw[i] = nw;
  out_emb[i]  = nw / (ncs + VQ_EPS);
  if (d == 0) out_cs[k] = ncs;
  if (i == 0) {
    out_scalars[0] = VQ_CC * scal[3] / (float)((size_t)VQ_N * VQ_D);  // loss
    out_scalars[1] = expf(scal[1]);                                   // perplexity
    out_scalars[2] = scal[2] / (float)VQ_K;                           // usage
  }
}

// ---------------------------------------------------------------- launcher
extern "C" void kernel_launch(void* const* d_in, const int* in_sizes, int n_in,
                              void* d_out, int out_size, void* d_ws, size_t ws_size,
                              hipStream_t stream) {
  const float* z      = (const float*)d_in[0];   // [32,64,64,64]
  const float* emb    = (const float*)d_in[1];   // [1024,64]
  const float* ema_cs = (const float*)d_in[2];   // [1024]
  const float* ema_w  = (const float*)d_in[3];   // [1024,64]

  float* out = (float*)d_out;
  float* o_zq   = out;                 // 8388608
  float* o_scal = out + 8388608;       // loss, perplexity, usage
  float* o_emb  = out + 8388611;       // 65536
  float* o_cs   = out + 8454147;       // 1024
  float* o_emaw = out + 8455171;       // 65536
  float* o_idx  = out + 8520707;       // 131072 (as float)

  char* ws = (char*)d_ws;
  _Float16* bpack = (_Float16*)ws;                // 131072 B
  float* cnp    = (float*)(ws + 131072);          //   4096 B
  int*   idxw   = (int*)  (ws + 135168);          // 524288 B
  float* counts = (float*)(ws + 659456);          //   4096 B
  float* dw     = (float*)(ws + 663552);          // 262144 B
  float* scal   = (float*)(ws + 925696);          //     64 B

  vq_zero_kernel   <<<256,  256, 0, stream>>>(counts, dw, scal);
  vq_prep_kernel   <<<256,  256, 0, stream>>>(emb, bpack, cnp);
  vq_argmin_kernel <<<512,  256, 0, stream>>>(z, bpack, cnp, idxw, o_idx);
  vq_scatter_kernel<<<512,  256, 0, stream>>>(z, idxw, counts, dw);
  vq_zq_loss_kernel<<<512,  256, 0, stream>>>(z, emb, idxw, o_zq, &scal[3]);
  vq_stats_kernel  <<<1,   1024, 0, stream>>>(ema_cs, counts, scal);
  vq_final_kernel  <<<256,  256, 0, stream>>>(ema_cs, ema_w, counts, dw, scal,
                                              o_emb, o_cs, o_emaw, o_scal);
  (void)in_sizes; (void)n_in; (void)out_size; (void)ws_size;
}
